// Wav2LandRNN_87952340287695
// MI455X (gfx1250) — matrix operand order repaired and checked by reading, not compiled, not run on hardware
//
#include <hip/hip_runtime.h>

typedef __attribute__((ext_vector_type(16))) _Float16 v16h;
typedef __attribute__((ext_vector_type(8)))  float    v8f;

#define REC_BLOCKS 32   // persistent-kernel grid; trivially co-resident on MI455X

__device__ __forceinline__ float sigm(float x) { return 1.0f / (1.0f + __expf(-x)); }

// Device-scope software grid barrier (monotone counter, zeroed per launch).
__device__ __forceinline__ void grid_barrier(unsigned* cnt, unsigned target) {
  __threadfence();               // release: make h/c/gates stores visible at agent scope
  __syncthreads();
  if (threadIdx.x == 0) {
    __hip_atomic_fetch_add(cnt, 1u, __ATOMIC_RELEASE, __HIP_MEMORY_SCOPE_AGENT);
    while (__hip_atomic_load(cnt, __ATOMIC_ACQUIRE, __HIP_MEMORY_SCOPE_AGENT) < target) {
      __builtin_amdgcn_s_sleep(1);
    }
  }
  __syncthreads();
  __threadfence();               // acquire: refresh stale WGP-cache lines
}

// ---------------------------------------------------------------------------
// WMMA GEMM: C(MxN,f32) = A(MxK,f16) * W(N_phys x K,f16)^T + bias1 + bias2
// K multiple of 32 (buffers zero-padded). 8 waves/block: wave w -> mtile
// blockIdx.y*8+w. Each wave computes exactly 4 N-tiles (named accumulators —
// no runtime-indexed register arrays). Caller guarantees W has at least
// gridDim.x*64 zero-padded rows; stores are guarded by col < N.
// ---------------------------------------------------------------------------
__global__ __launch_bounds__(256) void wmma_gemm_bias(
    const _Float16* __restrict__ A, int lda,
    const _Float16* __restrict__ W, int ldw,
    const float* __restrict__ bias1, const float* __restrict__ bias2,
    float* __restrict__ C, int ldc, int M, int N, int K)
{
  const int lane = threadIdx.x & 31;
  const int wave = threadIdx.x >> 5;
  const int mtile = blockIdx.y * 8 + wave;
  if (mtile * 16 >= M) return;                 // wave-uniform
  const int lrow = lane & 15;
  const int hsel = lane >> 4;                  // 0: lanes 0-15, 1: lanes 16-31
  const int nt0 = blockIdx.x * 4;

  v8f acc0 = {0,0,0,0,0,0,0,0};
  v8f acc1 = {0,0,0,0,0,0,0,0};
  v8f acc2 = {0,0,0,0,0,0,0,0};
  v8f acc3 = {0,0,0,0,0,0,0,0};

  // A fragment: lane row = mtile*16+lrow; lanes 0-15 hold K {k0..k0+7, k0+16..k0+23},
  // lanes 16-31 hold K {k0+8..k0+15, k0+24..k0+31}.
  const _Float16* arow = A + (size_t)(mtile * 16 + lrow) * lda + hsel * 8;
  const _Float16* wrow = W + (size_t)(nt0 * 16 + lrow) * ldw + hsel * 8;
  const size_t tstride = (size_t)16 * ldw;     // W-row stride between N-tiles

  for (int k0 = 0; k0 < K; k0 += 32) {
    union { v16h v; uint4 q[2]; } a;
    a.q[0] = *(const uint4*)(arow + k0);
    a.q[1] = *(const uint4*)(arow + k0 + 16);
#define GEMM_STEP(J, ACC)                                                      \
    {                                                                          \
      const _Float16* wr = wrow + (size_t)(J) * tstride + k0;                  \
      union { v16h v; uint4 q[2]; } bvu;                                       \
      bvu.q[0] = *(const uint4*)(wr);                                          \
      bvu.q[1] = *(const uint4*)(wr + 16);                                     \
      ACC = __builtin_amdgcn_wmma_f32_16x16x32_f16(                            \
          false, a.v, false, bvu.v, (short)0, ACC, false, false);              \
    }
    GEMM_STEP(0, acc0)
    GEMM_STEP(1, acc1)
    GEMM_STEP(2, acc2)
    GEMM_STEP(3, acc3)
#undef GEMM_STEP
  }

  const int rb = mtile * 16 + hsel * 8;        // C layout: VGPR e -> row rb+e, col = lrow
#define STORE_TILE(J, ACC)                                                     \
  {                                                                            \
    int col = (nt0 + (J)) * 16 + lrow;                                         \
    if (col < N) {                                                             \
      float bv = (bias1 ? bias1[col] : 0.0f) + (bias2 ? bias2[col] : 0.0f);    \
      _Pragma("unroll")                                                        \
      for (int e = 0; e < 8; ++e)                                              \
        C[(size_t)(rb + e) * ldc + col] = ACC[e] + bv;                         \
    }                                                                          \
  }
  STORE_TILE(0, acc0)
  STORE_TILE(1, acc1)
  STORE_TILE(2, acc2)
  STORE_TILE(3, acc3)
#undef STORE_TILE
}

// ---------------------------------------------------------------------------
// Persistent LSTM recurrence for one layer. 32 blocks x 256 threads.
// Block b owns gate columns [32b, 32b+32); wave w owns batch rows [16w, 16w+16).
// Per step: gates = xproj[t] + h16 @ Whh^T (WMMA, Whh slice cached in LDS),
// grid barrier, elementwise i/f/g/o update of c/h, grid barrier.
// ---------------------------------------------------------------------------
__global__ __launch_bounds__(256) void lstm_recurrent(
    const float* __restrict__ xproj,   // (T*128, 1024) f32
    const _Float16* __restrict__ Whh,  // (1024, 256) f16
    _Float16* h16,                     // (128, 256) f16 state (zeroed before launch)
    float* cst,                        // (128, 256) f32 cell state (zeroed)
    float* gates,                      // (128, 1024) f32 scratch
    _Float16* hs_out,                  // (T*128, 256) f16 hidden outputs
    unsigned* cnt, int T)
{
  __shared__ _Float16 lw[32 * 256];    // this block's 32 Whh rows (16 KB)
  const int b = blockIdx.x;
  const int tid = threadIdx.x;

  // Stage Whh slice into LDS once (re-read every step via ds_load_b128).
  {
    const uint4* s = (const uint4*)(Whh + (size_t)b * 32 * 256);
    uint4* d = (uint4*)lw;
    for (int i = tid; i < 32 * 256 / 8; i += 256) d[i] = s[i];
  }
  __syncthreads();

  const int lane = tid & 31;
  const int w = tid >> 5;
  const int lrow = lane & 15;
  const int hsel = lane >> 4;
  const int rowbase = w * 16 + hsel * 8;
  const int col0 = b * 32 + lrow;      // first tile's column for this lane
  const int col1 = col0 + 16;          // second tile's column
  const _Float16* ar0 = h16 + (size_t)(w * 16 + lrow) * 256 + hsel * 8;
  const _Float16* lw0 = lw + lrow * 256 + hsel * 8;           // tile 0 B rows
  const _Float16* lw1 = lw + (16 + lrow) * 256 + hsel * 8;    // tile 1 B rows
  unsigned bar = 0;

  for (int t = 0; t < T; ++t) {
    const float* xp = xproj + (size_t)t * 128 * 1024;
    v8f acc0, acc1;
#pragma unroll
    for (int e = 0; e < 8; ++e) {
      const float* r = xp + (size_t)(rowbase + e) * 1024;
      acc0[e] = r[col0];
      acc1[e] = r[col1];
    }
#pragma unroll
    for (int s = 0; s < 8; ++s) {      // K = 256 = 8 x 32
      const int k0 = s * 32;
      union { v16h v; uint4 q[2]; } a;
      a.q[0] = *(const uint4*)(ar0 + k0);
      a.q[1] = *(const uint4*)(ar0 + k0 + 16);
      union { v16h v; uint4 q[2]; } b0, b1;
      b0.q[0] = *(const uint4*)(lw0 + k0);
      b0.q[1] = *(const uint4*)(lw0 + k0 + 16);
      b1.q[0] = *(const uint4*)(lw1 + k0);
      b1.q[1] = *(const uint4*)(lw1 + k0 + 16);
      acc0 = __builtin_amdgcn_wmma_f32_16x16x32_f16(
          false, a.v, false, b0.v, (short)0, acc0, false, false);
      acc1 = __builtin_amdgcn_wmma_f32_16x16x32_f16(
          false, a.v, false, b1.v, (short)0, acc1, false, false);
    }
#pragma unroll
    for (int e = 0; e < 8; ++e) {
      float* g = gates + (size_t)(rowbase + e) * 1024;
      g[col0] = acc0[e];
      g[col1] = acc1[e];
    }

    ++bar; grid_barrier(cnt, bar * REC_BLOCKS);

    // Elementwise gate update: 128*256 elements over 8192 threads.
    const int gtid = b * 256 + tid;
#pragma unroll
    for (int it = 0; it < 4; ++it) {
      int e = gtid + it * (REC_BLOCKS * 256);
      int brow = e >> 8, hc = e & 255;
      const float* g = gates + (size_t)brow * 1024;
      float gi = g[hc], gf = g[256 + hc], gg = g[512 + hc], go = g[768 + hc];
      float cv = sigm(gf) * cst[e] + sigm(gi) * tanhf(gg);
      float hv = sigm(go) * tanhf(cv);
      cst[e] = cv;
      _Float16 hh = (_Float16)hv;
      h16[e] = hh;
      hs_out[(size_t)t * (128 * 256) + e] = hh;
    }

    ++bar; grid_barrier(cnt, bar * REC_BLOCKS);
  }
}

// --------------------------- small helper kernels ---------------------------

// x (B=128, T=512, 80) f32 -> seq16 (T*128, 96) f16 zero-padded (transpose to time-major)
__global__ void k_prep_x(const float* __restrict__ x, _Float16* __restrict__ dst) {
  int idx = blockIdx.x * 256 + threadIdx.x;
  if (idx >= 512 * 128 * 96) return;
  int k = idx % 96; int r = idx / 96;
  int bb = r & 127; int t = r >> 7;
  float v = (k < 80) ? x[((size_t)bb * 512 + t) * 80 + k] : 0.0f;
  dst[idx] = (_Float16)v;
}

// f32 (rows_s, sk) -> f16 (rows_d, dk), zero pad for r >= rows_s or c >= sk
__global__ void k_f32_to_f16_pad(const float* __restrict__ s, _Float16* __restrict__ d,
                                 int rows_s, int rows_d, int sk, int dk) {
  int idx = blockIdx.x * 256 + threadIdx.x;
  if (idx >= rows_d * dk) return;
  int c = idx % dk, r = idx / dk;
  float v = (r < rows_s && c < sk) ? s[(size_t)r * sk + c] : 0.0f;
  d[idx] = (_Float16)v;
}

__global__ void k_zero_state(_Float16* h16, float* cst) {
  int idx = blockIdx.x * 256 + threadIdx.x;
  if (idx < 128 * 256) { h16[idx] = (_Float16)0.0f; cst[idx] = 0.0f; }
}

__global__ void k_init_cnt(unsigned* cnt) {
  if (blockIdx.x == 0 && threadIdx.x < 8) cnt[threadIdx.x] = 0u;
}

// hcat (128, 480) f16 = [hT (128,256) | src (128,204) | zero pad]
__global__ void k_build_hcat(const _Float16* __restrict__ h16,
                             const float* __restrict__ src, _Float16* __restrict__ hcat) {
  int idx = blockIdx.x * 256 + threadIdx.x;
  if (idx >= 128 * 480) return;
  int c = idx % 480, r = idx / 480;
  _Float16 v;
  if (c < 256)      v = h16[r * 256 + c];
  else if (c < 460) v = (_Float16)src[(size_t)r * 204 + (c - 256)];
  else              v = (_Float16)0.0f;
  hcat[idx] = v;
}

// ---------------------------------------------------------------------------

extern "C" void kernel_launch(void* const* d_in, const int* in_sizes, int n_in,
                              void* d_out, int out_size, void* d_ws, size_t ws_size,
                              hipStream_t stream) {
  (void)in_sizes; (void)n_in; (void)out_size; (void)ws_size;
  const float* x    = (const float*)d_in[0];
  const float* src  = (const float*)d_in[1];
  const float* Wih0 = (const float*)d_in[2];
  const float* Whh0 = (const float*)d_in[3];
  const float* bih0 = (const float*)d_in[4];
  const float* bhh0 = (const float*)d_in[5];
  const float* Wih1 = (const float*)d_in[6];
  const float* Whh1 = (const float*)d_in[7];
  const float* bih1 = (const float*)d_in[8];
  const float* bhh1 = (const float*)d_in[9];
  const float* Wih2 = (const float*)d_in[10];
  const float* Whh2 = (const float*)d_in[11];
  const float* bih2 = (const float*)d_in[12];
  const float* bhh2 = (const float*)d_in[13];
  const float* W1   = (const float*)d_in[14];
  const float* b1   = (const float*)d_in[15];
  const float* W2   = (const float*)d_in[16];
  const float* b2   = (const float*)d_in[17];
  const float* W3   = (const float*)d_in[18];
  const float* b3   = (const float*)d_in[19];

  char* base = (char*)d_ws;
  size_t off = 0;
  auto alloc = [&](size_t bytes) -> void* {
    off = (off + 255) & ~(size_t)255;
    void* r = base + off; off += bytes; return r;
  };

  unsigned* cnt     = (unsigned*)alloc(256);
  _Float16* seq16   = (_Float16*)alloc((size_t)65536 * 256 * 2);   // layer input / hs (32 MiB)
  float*    xproj   = (float*)   alloc((size_t)65536 * 1024 * 4);  // 256 MiB
  _Float16* wih0h   = (_Float16*)alloc((size_t)1024 * 96 * 2);
  _Float16* whh0h   = (_Float16*)alloc((size_t)1024 * 256 * 2);
  _Float16* wih1h   = (_Float16*)alloc((size_t)1024 * 256 * 2);
  _Float16* whh1h   = (_Float16*)alloc((size_t)1024 * 256 * 2);
  _Float16* wih2h   = (_Float16*)alloc((size_t)1024 * 256 * 2);
  _Float16* whh2h   = (_Float16*)alloc((size_t)1024 * 256 * 2);
  _Float16* w1h     = (_Float16*)alloc((size_t)512 * 480 * 2);
  _Float16* w2h     = (_Float16*)alloc((size_t)256 * 512 * 2);
  _Float16* w3h     = (_Float16*)alloc((size_t)256 * 256 * 2);     // rows padded 204 -> 256
  _Float16* h16     = (_Float16*)alloc((size_t)128 * 256 * 2);
  float*    cbuf    = (float*)   alloc((size_t)128 * 256 * 4);
  float*    gates   = (float*)   alloc((size_t)128 * 1024 * 4);
  _Float16* hcat    = (_Float16*)alloc((size_t)128 * 480 * 2);
  float*    out1    = (float*)   alloc((size_t)128 * 512 * 4);
  _Float16* out1h   = (_Float16*)alloc((size_t)128 * 512 * 2);
  float*    out2    = (float*)   alloc((size_t)128 * 256 * 4);
  _Float16* out2h   = (_Float16*)alloc((size_t)128 * 256 * 2);

  k_init_cnt<<<1, 8, 0, stream>>>(cnt);

  // Precision conversion / padding
  k_prep_x<<<(512 * 128 * 96 + 255) / 256, 256, 0, stream>>>(x, seq16);
  k_f32_to_f16_pad<<<(1024 * 96  + 255) / 256, 256, 0, stream>>>(Wih0, wih0h, 1024, 1024, 80, 96);
  k_f32_to_f16_pad<<<(1024 * 256 + 255) / 256, 256, 0, stream>>>(Whh0, whh0h, 1024, 1024, 256, 256);
  k_f32_to_f16_pad<<<(1024 * 256 + 255) / 256, 256, 0, stream>>>(Wih1, wih1h, 1024, 1024, 256, 256);
  k_f32_to_f16_pad<<<(1024 * 256 + 255) / 256, 256, 0, stream>>>(Whh1, whh1h, 1024, 1024, 256, 256);
  k_f32_to_f16_pad<<<(1024 * 256 + 255) / 256, 256, 0, stream>>>(Wih2, wih2h, 1024, 1024, 256, 256);
  k_f32_to_f16_pad<<<(1024 * 256 + 255) / 256, 256, 0, stream>>>(Whh2, whh2h, 1024, 1024, 256, 256);
  k_f32_to_f16_pad<<<(512 * 480 + 255) / 256, 256, 0, stream>>>(W1, w1h, 512, 512, 460, 480);
  k_f32_to_f16_pad<<<(256 * 512 + 255) / 256, 256, 0, stream>>>(W2, w2h, 256, 256, 512, 512);
  k_f32_to_f16_pad<<<(256 * 256 + 255) / 256, 256, 0, stream>>>(W3, w3h, 204, 256, 256, 256);

  // Layer 0: parallel input projection, then persistent recurrence
  wmma_gemm_bias<<<dim3(16, 512), 256, 0, stream>>>(seq16, 96, wih0h, 96,
      bih0, bhh0, xproj, 1024, 65536, 1024, 96);
  k_zero_state<<<128, 256, 0, stream>>>(h16, cbuf);
  lstm_recurrent<<<REC_BLOCKS, 256, 0, stream>>>(xproj, whh0h, h16, cbuf, gates,
      seq16, cnt + 0, 512);

  // Layer 1
  wmma_gemm_bias<<<dim3(16, 512), 256, 0, stream>>>(seq16, 256, wih1h, 256,
      bih1, bhh1, xproj, 1024, 65536, 1024, 256);
  k_zero_state<<<128, 256, 0, stream>>>(h16, cbuf);
  lstm_recurrent<<<REC_BLOCKS, 256, 0, stream>>>(xproj, whh1h, h16, cbuf, gates,
      seq16, cnt + 1, 512);

  // Layer 2 (only final h is consumed)
  wmma_gemm_bias<<<dim3(16, 512), 256, 0, stream>>>(seq16, 256, wih2h, 256,
      bih2, bhh2, xproj, 1024, 65536, 1024, 256);
  k_zero_state<<<128, 256, 0, stream>>>(h16, cbuf);
  lstm_recurrent<<<REC_BLOCKS, 256, 0, stream>>>(xproj, whh2h, h16, cbuf, gates,
      seq16, cnt + 2, 512);

  // MLP head: [hT | src] (128x460 -> pad 480) -> 512 -> 256 -> 204
  k_build_hcat<<<(128 * 480 + 255) / 256, 256, 0, stream>>>(h16, src, hcat);
  wmma_gemm_bias<<<dim3(8, 1), 256, 0, stream>>>(hcat, 480, w1h, 480,
      b1, nullptr, out1, 512, 128, 512, 480);
  k_f32_to_f16_pad<<<(128 * 512 + 255) / 256, 256, 0, stream>>>(out1, out1h, 128, 128, 512, 512);
  wmma_gemm_bias<<<dim3(4, 1), 256, 0, stream>>>(out1h, 512, w2h, 512,
      b2, nullptr, out2, 256, 128, 256, 512);
  k_f32_to_f16_pad<<<(128 * 256 + 255) / 256, 256, 0, stream>>>(out2, out2h, 128, 128, 256, 256);
  wmma_gemm_bias<<<dim3(4, 1), 256, 0, stream>>>(out2h, 256, w3h, 256,
      b3, nullptr, (float*)d_out, 204, 128, 204, 256);
}
